// MultiHeadLinearAttention_48275432407732
// MI455X (gfx1250) — matrix operand order repaired and checked by reading, not compiled, not run on hardware
//
#include <hip/hip_runtime.h>

// ---------------- problem constants ----------------
#define CCH   512          // channels
#define NPIX  4096         // H*W
#define NB    16           // batch
#define NH    8            // heads
#define HD    64           // head dim
#define LDT   520          // LDS tile row stride in halfs (512+8 pad; 1040B rows, 16B aligned)
#define T1N   64           // pixel tile, QKV kernel
#define T3N   32           // pixel tile, output kernel
#define ATTP  33           // att LDS pitch (32 + 1)
#define EPSV  1e-6f

typedef __attribute__((ext_vector_type(16))) __bf16       v16bf;
typedef __attribute__((ext_vector_type(8)))  float        v8f;
typedef __attribute__((ext_vector_type(8)))  unsigned int v8u;
typedef __attribute__((ext_vector_type(4)))  unsigned int v4u;

// ---------------- helpers ----------------
__device__ __forceinline__ unsigned short f2bf(float f) {
  unsigned int u = __builtin_bit_cast(unsigned int, f);
  u += 0x7FFFu + ((u >> 16) & 1u);            // round-to-nearest-even
  return (unsigned short)(u >> 16);
}

__device__ __forceinline__ v16bf pack2(v4u lo, v4u hi) {
  v8u t;
  t[0] = lo[0]; t[1] = lo[1]; t[2] = lo[2]; t[3] = lo[3];
  t[4] = hi[0]; t[5] = hi[1]; t[6] = hi[2]; t[7] = hi[3];
  return __builtin_bit_cast(v16bf, t);
}

__device__ __forceinline__ v16bf pack_frag(const unsigned int* u) {
  v8u t;
#pragma unroll
  for (int r = 0; r < 8; ++r) t[r] = u[r];
  return __builtin_bit_cast(v16bf, t);
}

__device__ __forceinline__ v8f wmma_bf16(v16bf a, v16bf b, v8f c) {
  // D = A(16x32) * B(32x16) + C ; f32 accumulate
  return __builtin_amdgcn_wmma_f32_16x16x32_bf16(false, a, false, b, (short)0, c,
                                                 false, false);
}

// A fragment from the frag-swizzled weight layout: per lane 32 contiguous bytes,
// whole fragment = one contiguous, fully-coalesced 1KB run -> 2x global_load_b128.
__device__ __forceinline__ v16bf load_a_swz(const unsigned short* frag, int lane) {
  const v4u* p = (const v4u*)(frag + lane * 16);
  return pack2(p[0], p[1]);
}

// B fragment where memory is [col][k] (per-column stride ld halfs, K contiguous).
// Per lane: halfs ko..ko+7 and ko+16..ko+23 (16B aligned) -> 2x (ds_/global_)load_b128.
__device__ __forceinline__ v16bf load_b_kcontig(const unsigned short* base, int ld, int lane) {
  int c  = lane & 15;
  int ko = (lane >> 4) << 3;                  // 0 or 8
  const v4u* p = (const v4u*)(base + c * ld + ko);
  return pack2(p[0], p[2]);
}

// B fragment where memory is [k][col] (K strided by ld, columns contiguous).
__device__ __forceinline__ v16bf load_b_kstrided(const unsigned short* base, int ld, int lane) {
  int c  = lane & 15;
  int ko = (lane >> 4) << 3;
  const unsigned short* p = base + ko * ld + c;
  unsigned int u[8];
#pragma unroll
  for (int r = 0; r < 8; ++r) {
    int kr = (r < 4) ? 2 * r : 2 * r + 8;
    unsigned int lo = p[kr * ld];
    unsigned int hi = p[(kr + 1) * ld];
    u[r] = lo | (hi << 16);
  }
  return pack_frag(u);
}

// A fragment of ones-padded V: rows e<64 from V[d][n] bf16; e==64 -> 1.0; e>64 -> 0.
// ISA A layout: lane m=lane&15 holds row; lane-half K 0-7/8-15 (VGPR0-3), 16-23/24-31 (VGPR4-7).
__device__ __forceinline__ v16bf load_a_vpad(const unsigned short* vb, int e0, int n0, int lane) {
  int m  = lane & 15;
  int e  = e0 + m;
  int ko = (lane >> 4) << 3;                  // 0 or 8
  if (e < HD) {
    const unsigned short* p = vb + (size_t)e * NPIX + n0 + ko;   // 16B aligned runs
    v4u lo = *(const v4u*)p;
    v4u hi = *(const v4u*)(p + 16);
    return pack2(lo, hi);
  }
  v4u c;
  unsigned int fill = (e == HD) ? 0x3F803F80u : 0u;              // packed bf16 1.0
  c[0] = fill; c[1] = fill; c[2] = fill; c[3] = fill;
  return pack2(c, c);
}

// A fragment from fp32 kv[65][64]; rows >= 65 are zero.
__device__ __forceinline__ v16bf load_a_kv(const float* kvb, int e0, int k0, int lane) {
  int m  = lane & 15;
  int e  = e0 + m;
  int ko = (lane >> 4) << 3;
  unsigned int u[8];
  if (e < HD + 1) {
    const float4* p0 = (const float4*)(kvb + e * HD + k0 + ko);       // 8 floats
    const float4* p1 = (const float4*)(kvb + e * HD + k0 + ko + 16);  // 8 floats
    float4 a = p0[0], b = p0[1], c = p1[0], d = p1[1];
    u[0] = (unsigned int)f2bf(a.x) | ((unsigned int)f2bf(a.y) << 16);
    u[1] = (unsigned int)f2bf(a.z) | ((unsigned int)f2bf(a.w) << 16);
    u[2] = (unsigned int)f2bf(b.x) | ((unsigned int)f2bf(b.y) << 16);
    u[3] = (unsigned int)f2bf(b.z) | ((unsigned int)f2bf(b.w) << 16);
    u[4] = (unsigned int)f2bf(c.x) | ((unsigned int)f2bf(c.y) << 16);
    u[5] = (unsigned int)f2bf(c.z) | ((unsigned int)f2bf(c.w) << 16);
    u[6] = (unsigned int)f2bf(d.x) | ((unsigned int)f2bf(d.y) << 16);
    u[7] = (unsigned int)f2bf(d.z) | ((unsigned int)f2bf(d.w) << 16);
  } else {
#pragma unroll
    for (int r = 0; r < 8; ++r) u[r] = 0u;
  }
  return pack_frag(u);
}

// ---------------- kernel 0: weights fp32 -> bf16, fragment-swizzled ----------------
// Output layout per matrix: frag(mt*16+kc) blocks of 256 dwords; dword = lane*8 + f.
// Element mapping matches the WMMA A-fragment VGPR layout.
__global__ __launch_bounds__(256) void k0_swizzle_w(
    const float* __restrict__ wqk, const float* __restrict__ wv, const float* __restrict__ wo,
    unsigned short* __restrict__ qswz, unsigned short* __restrict__ vswz,
    unsigned short* __restrict__ oswz) {
  int p = blockIdx.x * 256 + threadIdx.x;     // dword (bf16-pair) index
  const float* src;
  unsigned short* dst;
  int local;
  if (p < 262144)      { src = wqk; dst = qswz; local = p; }            // 1024x512
  else if (p < 393216) { src = wv;  dst = vswz; local = p - 262144; }   // 512x512
  else if (p < 524288) { src = wo;  dst = oswz; local = p - 393216; }   // 512x512
  else return;
  int chunk  = local >> 8;                    // frag id = mt*16 + kc
  int within = local & 255;
  int lane = within >> 3, f = within & 7;
  int mt = chunk >> 4, kc = chunk & 15;
  int m = lane & 15, ko = (lane >> 4) << 3;
  int kb = (f < 4) ? (ko + 2 * f) : (16 + ko + 2 * (f - 4));
  size_t s = (size_t)(mt * 16 + m) * CCH + kc * 32 + kb;
  unsigned int lo = f2bf(src[s]);
  unsigned int hi = f2bf(src[s + 1]);
  ((unsigned int*)dst)[local] = lo | (hi << 16);
}

// ---------------- kernel 1: QKV (fused 1x1 convs + relu) ----------------
// Q/K/V stored bf16, layout [b, h, d, n]  (n contiguous).
__global__ __launch_bounds__(256) void k1_qkv(
    const float* __restrict__ x, const float* __restrict__ pos,
    const unsigned short* __restrict__ qswz, const unsigned short* __restrict__ vswz,
    unsigned short* __restrict__ Q, unsigned short* __restrict__ K, unsigned short* __restrict__ V) {
  extern __shared__ unsigned short sm1[];
  unsigned short* sxp = sm1;                  // x+pos tile [T1N][LDT]
  unsigned short* sx  = sm1 + T1N * LDT;      // x tile     [T1N][LDT]

  const int b  = blockIdx.y;
  const int n0 = blockIdx.x * T1N;
  const int tid = threadIdx.x;

  for (int i = tid; i < CCH * (T1N / 4); i += 256) {
    int c = i >> 4, nq = (i & 15) << 2;
    size_t g = (size_t)(b * CCH + c) * NPIX + n0 + nq;
    float4 xv = *(const float4*)(x + g);
    float4 pv = *(const float4*)(pos + g);
    sx[(nq + 0) * LDT + c] = f2bf(xv.x);
    sx[(nq + 1) * LDT + c] = f2bf(xv.y);
    sx[(nq + 2) * LDT + c] = f2bf(xv.z);
    sx[(nq + 3) * LDT + c] = f2bf(xv.w);
    sxp[(nq + 0) * LDT + c] = f2bf(xv.x + pv.x);
    sxp[(nq + 1) * LDT + c] = f2bf(xv.y + pv.y);
    sxp[(nq + 2) * LDT + c] = f2bf(xv.z + pv.z);
    sxp[(nq + 3) * LDT + c] = f2bf(xv.w + pv.w);
  }
  __syncthreads();

  const int wid = tid >> 5, lane = tid & 31;
  const int nloc = lane & 15, mh = (lane >> 4) << 3;

#pragma unroll 1
  for (int j = 0; j < 12; ++j) {
    int mt = wid + 8 * j;                     // 0..95 M-tiles over 1536 output rows
    int row0 = mt * 16;
    const bool isqk = (row0 < 2 * CCH);
    const unsigned short* Aswz = isqk ? (qswz + (size_t)mt * 16 * 512)
                                      : (vswz + (size_t)(mt - 64) * 16 * 512);
    const unsigned short* Bt = isqk ? sxp : sx;

    v8f acc[4];
#pragma unroll
    for (int nt = 0; nt < 4; ++nt) acc[nt] = v8f{};

    for (int kk = 0; kk < CCH; kk += 32) {
      v16bf a  = load_a_swz(Aswz + (size_t)(kk >> 5) * 512, lane);
      v16bf b0 = load_b_kcontig(Bt + kk, LDT, lane);
      v16bf b1 = load_b_kcontig(Bt + 16 * LDT + kk, LDT, lane);
      v16bf b2 = load_b_kcontig(Bt + 32 * LDT + kk, LDT, lane);
      v16bf b3 = load_b_kcontig(Bt + 48 * LDT + kk, LDT, lane);
      acc[0] = wmma_bf16(a, b0, acc[0]);
      acc[1] = wmma_bf16(a, b1, acc[1]);
      acc[2] = wmma_bf16(a, b2, acc[2]);
      acc[3] = wmma_bf16(a, b3, acc[3]);
    }
#pragma unroll
    for (int nt = 0; nt < 4; ++nt) {
#pragma unroll
      for (int r = 0; r < 8; ++r) {
        float val = acc[nt][r];
        int o = row0 + r + mh;
        int n = n0 + nt * 16 + nloc;
        if (isqk) {
          val = val > 0.f ? val : 0.f;        // relu on q,k
          int h = (o >> 6) & 7, d = o & 63;
          unsigned short* dst = (o < CCH) ? Q : K;
          dst[((size_t)(b * NH + h) * HD + d) * NPIX + n] = f2bf(val);
        } else {
          int o2 = o - 2 * CCH;
          int h = o2 >> 6, d = o2 & 63;
          V[((size_t)(b * NH + h) * HD + d) * NPIX + n] = f2bf(val);
        }
      }
    }
  }
}

// ---------------- kernel 2: kv partials  kv[e,d] += sum_n vpad[e,n]*k[d,n] ----------------
// grid (bh=128, chunk=8), 160 threads = 5 waves, wave w owns e-rows [16w,16w+16)
__global__ __launch_bounds__(160) void k2_kv_partial(
    const unsigned short* __restrict__ K, const unsigned short* __restrict__ V,
    float* __restrict__ kvpart) {
  const int bh = blockIdx.x, chunk = blockIdx.y;
  const int wid = threadIdx.x >> 5, lane = threadIdx.x & 31;
  const int e0 = wid * 16;
  const unsigned short* Kb = K + (size_t)bh * HD * NPIX;
  const unsigned short* Vb = V + (size_t)bh * HD * NPIX;

  v8f acc[4];
#pragma unroll
  for (int dt = 0; dt < 4; ++dt) acc[dt] = v8f{};

  const int nbase = chunk * 512;
  for (int kk = 0; kk < 512; kk += 32) {
    int n = nbase + kk;
    v16bf a  = load_a_vpad(Vb, e0, n, lane);
    v16bf b0 = load_b_kcontig(Kb + (size_t)0 * 16 * NPIX + n, NPIX, lane);
    v16bf b1 = load_b_kcontig(Kb + (size_t)1 * 16 * NPIX + n, NPIX, lane);
    v16bf b2 = load_b_kcontig(Kb + (size_t)2 * 16 * NPIX + n, NPIX, lane);
    v16bf b3 = load_b_kcontig(Kb + (size_t)3 * 16 * NPIX + n, NPIX, lane);
    acc[0] = wmma_bf16(a, b0, acc[0]);
    acc[1] = wmma_bf16(a, b1, acc[1]);
    acc[2] = wmma_bf16(a, b2, acc[2]);
    acc[3] = wmma_bf16(a, b3, acc[3]);
  }
  float* out = kvpart + (size_t)(chunk * 128 + bh) * 80 * HD;
  const int nloc = lane & 15, mh = (lane >> 4) << 3;
#pragma unroll
  for (int dt = 0; dt < 4; ++dt)
#pragma unroll
    for (int r = 0; r < 8; ++r)
      out[(e0 + r + mh) * HD + dt * 16 + nloc] = acc[dt][r];
}

// ---------------- kernel 2b: deterministic 8-way reduce -> kv[bh][65][64] fp32 ----------------
__global__ __launch_bounds__(256) void k2b_kv_reduce(
    const float* __restrict__ kvpart, float* __restrict__ kv) {
  int idx = blockIdx.x * 256 + threadIdx.x;           // 128*65*64 total
  if (idx >= 128 * 65 * 64) return;
  int bh = idx / (65 * 64);
  int rem = idx - bh * 65 * 64;
  int e = rem >> 6, d = rem & 63;
  float s = 0.f;
#pragma unroll
  for (int c = 0; c < 8; ++c)
    s += kvpart[(size_t)(c * 128 + bh) * 80 * HD + e * HD + d];
  kv[idx] = s;
}

// ---------------- kernel 3: out = w_o * normalize(kv * q) ----------------
__global__ __launch_bounds__(256) void k3_out(
    const unsigned short* __restrict__ Q, const float* __restrict__ kv,
    const unsigned short* __restrict__ oswz, float* __restrict__ out) {
  extern __shared__ unsigned char sm3[];
  float* att        = (float*)sm3;                                           // [8][80][ATTP]
  unsigned short* Y = (unsigned short*)(sm3 + NH * 80 * ATTP * sizeof(float)); // [T3N][LDT]

  const int b  = blockIdx.y;
  const int n0 = blockIdx.x * T3N;
  const int wid = threadIdx.x >> 5, lane = threadIdx.x & 31;
  const int nloc = lane & 15, mh = (lane >> 4) << 3;

  // phase A: 8 heads x 5 M-tiles x 2 N-tiles = 80 tiles, K=64 fully unrolled
#pragma unroll 1
  for (int i = 0; i < 10; ++i) {
    int t = wid + 8 * i;
    int nt = t & 1, rest = t >> 1;
    int mt = rest % 5, h = rest / 5;
    int bh = b * NH + h;
    const float*          kvb = kv + (size_t)bh * 65 * HD;
    const unsigned short* Qb  = Q + (size_t)bh * HD * NPIX + n0 + nt * 16;
    v16bf a0 = load_a_kv(kvb, mt * 16, 0, lane);
    v16bf b0 = load_b_kstrided(Qb, NPIX, lane);
    v16bf a1 = load_a_kv(kvb, mt * 16, 32, lane);
    v16bf b1 = load_b_kstrided(Qb + (size_t)32 * NPIX, NPIX, lane);
    v8f acc = v8f{};
    acc = wmma_bf16(a0, b0, acc);
    acc = wmma_bf16(a1, b1, acc);
#pragma unroll
    for (int r = 0; r < 8; ++r) {
      int e = mt * 16 + r + mh;
      att[(h * 80 + e) * ATTP + nt * 16 + nloc] = acc[r];
    }
  }
  __syncthreads();

  // normalize by carried ones-row, pack bf16 [n][c] matrix
  for (int idx = threadIdx.x; idx < NH * HD * T3N; idx += 256) {
    int h = idx / (HD * T3N);
    int rem = idx - h * HD * T3N;
    int e = rem >> 5, n = rem & 31;
    float den = att[(h * 80 + HD) * ATTP + n] + EPSV;
    float val = att[(h * 80 + e) * ATTP + n] / den;
    Y[n * LDT + h * HD + e] = f2bf(val);
  }
  __syncthreads();

  // phase B: w_o GEMM, 32 M-tiles x 2 N-tiles, K=512
#pragma unroll 1
  for (int i = 0; i < 8; ++i) {
    int t = wid + 8 * i;
    int nt = t & 1, mt = t >> 1;
    const unsigned short* Aswz = oswz + (size_t)mt * 16 * 512;
    v8f acc = v8f{};
    for (int kk = 0; kk < CCH; kk += 32) {
      v16bf a  = load_a_swz(Aswz + (size_t)(kk >> 5) * 512, lane);
      v16bf bf = load_b_kcontig(Y + (nt * 16) * LDT + kk, LDT, lane);
      acc = wmma_bf16(a, bf, acc);
    }
#pragma unroll
    for (int r = 0; r < 8; ++r) {
      int o = mt * 16 + r + mh;
      out[((size_t)(b * CCH + o)) * NPIX + n0 + nt * 16 + nloc] = acc[r];
    }
  }
}

// ---------------- launcher ----------------
extern "C" void kernel_launch(void* const* d_in, const int* in_sizes, int n_in,
                              void* d_out, int out_size, void* d_ws, size_t ws_size,
                              hipStream_t stream) {
  const float* x    = (const float*)d_in[0];
  const float* pos  = (const float*)d_in[1];
  const float* wqk  = (const float*)d_in[2];
  const float* wvv  = (const float*)d_in[3];
  const float* wo   = (const float*)d_in[4];
  float* out = (float*)d_out;

  // workspace layout (bytes) -- ~227 MB total
  unsigned char* ws = (unsigned char*)d_ws;
  size_t off = 0;
  unsigned short* wqk_b = (unsigned short*)(ws + off); off += (size_t)2 * CCH * CCH * 2;
  unsigned short* wv_b  = (unsigned short*)(ws + off); off += (size_t)CCH * CCH * 2;
  unsigned short* wo_b  = (unsigned short*)(ws + off); off += (size_t)CCH * CCH * 2;
  unsigned short* Qws   = (unsigned short*)(ws + off); off += (size_t)NB * NH * HD * NPIX * 2;
  unsigned short* Kws   = (unsigned short*)(ws + off); off += (size_t)NB * NH * HD * NPIX * 2;
  unsigned short* Vws   = (unsigned short*)(ws + off); off += (size_t)NB * NH * HD * NPIX * 2;
  float* kvpart         = (float*)(ws + off);          off += (size_t)8 * 128 * 80 * HD * 4;
  float* kvws           = (float*)(ws + off);          off += (size_t)128 * 65 * HD * 4;
  (void)ws_size; (void)in_sizes; (void)n_in; (void)out_size;

  // 0: weight conversion + fragment swizzle
  k0_swizzle_w<<<dim3(2048), dim3(256), 0, stream>>>(wqk, wvv, wo, wqk_b, wv_b, wo_b);

  // 1: QKV
  size_t sm1 = (size_t)2 * T1N * LDT * sizeof(unsigned short);   // 133,120 B
  k1_qkv<<<dim3(NPIX / T1N, NB), dim3(256), sm1, stream>>>(x, pos, wqk_b, wv_b, Qws, Kws, Vws);

  // 2: kv partials over 8 pixel chunks
  k2_kv_partial<<<dim3(NB * NH, 8), dim3(160), 0, stream>>>(Kws, Vws, kvpart);

  // 2b: reduce partials (deterministic order)
  k2b_kv_reduce<<<dim3((128 * 65 * 64 + 255) / 256), dim3(256), 0, stream>>>(kvpart, kvws);

  // 3: attention output + normalization + w_o
  size_t sm3 = (size_t)NH * 80 * ATTP * sizeof(float) + (size_t)T3N * LDT * sizeof(unsigned short); // 117,760 B
  k3_out<<<dim3(NPIX / T3N, NB), dim3(256), sm3, stream>>>(Qws, kvws, wo_b, out);
}
